// HiMambaBlock_12232066859081
// MI455X (gfx1250) — matrix operand order repaired and checked by reading, not compiled, not run on hardware
//
#include <hip/hip_runtime.h>
#include <hip/hip_bf16.h>

// ---------------------------------------------------------------------------
// HiMamba block for gfx1250 (MI455X). All GEMMs use v_wmma_f32_16x16x32_bf16.
// Problem constants from the reference.
// ---------------------------------------------------------------------------
#define DIMC   192
#define LSEQ   4096          // 64*64
#define BSZ    2
#define MROWS  (BSZ * LSEQ)  // 8192
#define NSTATE 16
#define DTRANK 12
#define DWDIM  384           // DIM*2
#define NCAT   224           // 192 (dt_eff) + 16 (B) + 16 (C)

typedef __attribute__((ext_vector_type(16))) __bf16 v16bf;
typedef __attribute__((ext_vector_type(8)))  float  v8f;

// ---------------------------------------------------------------------------
// Fragment loaders (ISA 7.12.2 VGPR layouts, 16-bit operands, wave32)
// A 16x32: lane half=lane>>4; regs 0..3 -> K = k0+half*8..+7 ; regs 4..7 -> +16
// B 32x16: lane half; regs 0..3 -> K = k0+half*8..+7 ; regs 4..7 -> +16 (same as A)
// ---------------------------------------------------------------------------
__device__ __forceinline__ v16bf ld_fragA(const __bf16* p /* &A[m*K + k0 + half*8] */) {
  union { uint4 q[2]; v16bf v; } u;
  u.q[0] = *(const uint4*)(p);
  u.q[1] = *(const uint4*)(p + 16);
  return u.v;
}
__device__ __forceinline__ v16bf ld_fragB(const __bf16* p /* &W[n*K + k0 + half*8] */) {
  union { uint4 q[2]; v16bf v; } u;
  u.q[0] = *(const uint4*)(p);
  u.q[1] = *(const uint4*)(p + 16);
  return u.v;
}

// ---------------------------------------------------------------------------
// WMMA GEMM:  C[M,N] = A[M,K] (bf16, row-major) * W[N,K]^T (bf16, row-major)
//             (+ bias[n]) (+= existing C if accumulate)
// Block = 128 threads = 4 waves. Each wave: 32 rows x 32 cols = 2x2 WMMA tiles.
// grid = (M/128, N/32). M%128==0, N%32==0, K%32==0 (all true here).
// ---------------------------------------------------------------------------
__global__ __launch_bounds__(128) void k_gemm_wmma(
    const __bf16* __restrict__ A, const __bf16* __restrict__ W,
    float* __restrict__ C, int M, int N, int K,
    const float* __restrict__ bias, int accumulate)
{
  const int lane = threadIdx.x & 31;
  const int wave = threadIdx.x >> 5;
  const int half = lane >> 4;
  const int l16  = lane & 15;
  const int row0 = blockIdx.x * 128 + wave * 32;
  const int col0 = blockIdx.y * 32;

  const __bf16* a0 = A + (size_t)(row0 + l16) * K + half * 8;
  const __bf16* a1 = a0 + (size_t)16 * K;
  const __bf16* b0 = W + (size_t)(col0 + l16) * K + half * 8;
  const __bf16* b1 = b0 + (size_t)16 * K;

  v8f acc00 = {}, acc01 = {}, acc10 = {}, acc11 = {};
  for (int k0 = 0; k0 < K; k0 += 32) {
    v16bf fa0 = ld_fragA(a0 + k0);
    v16bf fa1 = ld_fragA(a1 + k0);
    v16bf fb0 = ld_fragB(b0 + k0);
    v16bf fb1 = ld_fragB(b1 + k0);
    acc00 = __builtin_amdgcn_wmma_f32_16x16x32_bf16(false, fa0, false, fb0, (short)0, acc00, false, false);
    acc01 = __builtin_amdgcn_wmma_f32_16x16x32_bf16(false, fa0, false, fb1, (short)0, acc01, false, false);
    acc10 = __builtin_amdgcn_wmma_f32_16x16x32_bf16(false, fa1, false, fb0, (short)0, acc10, false, false);
    acc11 = __builtin_amdgcn_wmma_f32_16x16x32_bf16(false, fa1, false, fb1, (short)0, acc11, false, false);
  }

  // D layout: lanes 0-15 -> n=lane, M=reg; lanes 16-31 -> n=lane-16, M=reg+8
  v8f accs[2][2] = {{acc00, acc01}, {acc10, acc11}};
  for (int mi = 0; mi < 2; ++mi) {
    for (int ni = 0; ni < 2; ++ni) {
      const int n  = col0 + ni * 16 + l16;
      const float badd = bias ? bias[n] : 0.f;
      #pragma unroll
      for (int r = 0; r < 8; ++r) {
        const int m = row0 + mi * 16 + half * 8 + r;
        size_t idx = (size_t)m * N + n;
        float v = accs[mi][ni][r] + badd;
        if (accumulate) v += C[idx];
        C[idx] = v;
      }
    }
  }
}

// ---------------------------------------------------------------------------
// f32 -> bf16 copy
// ---------------------------------------------------------------------------
__global__ void k_cvt(const float* __restrict__ src, __bf16* __restrict__ dst, int n) {
  int i = blockIdx.x * 256 + threadIdx.x;
  if (i < n) dst[i] = (__bf16)src[i];
}

// Build concatenated projection weight [224,192]:
// rows 0..191  : dt_eff[e,d] = sum_r dt_w[e,r] * xproj[r,d]   (Δt folded)
// rows 192..223: xproj rows 12..43 (B then C)
__global__ void k_xcat(const float* __restrict__ xproj, const float* __restrict__ dtw,
                       __bf16* __restrict__ Wx) {
  int i = blockIdx.x * 256 + threadIdx.x;
  if (i >= NCAT * DIMC) return;
  int e = i / DIMC, d = i % DIMC;
  float v;
  if (e < DIMC) {
    v = 0.f;
    #pragma unroll
    for (int r = 0; r < DTRANK; ++r) v += dtw[e * DTRANK + r] * xproj[r * DIMC + d];
  } else {
    v = xproj[(DTRANK + (e - DIMC)) * DIMC + d];
  }
  Wx[i] = (__bf16)v;
}

// ---------------------------------------------------------------------------
// LayerNorm over channels; x [B,C,H,W] f32 -> XN [M,192] bf16 (BHWC rows)
// One wave per row; lane handles 6 strided channels.
// ---------------------------------------------------------------------------
__global__ __launch_bounds__(128) void k_layernorm(
    const float* __restrict__ x, const float* __restrict__ lnw,
    const float* __restrict__ lnb, __bf16* __restrict__ XN)
{
  int row  = blockIdx.x * 4 + (threadIdx.x >> 5);
  int lane = threadIdx.x & 31;
  int b = row >> 12, l = row & (LSEQ - 1);
  const float* xp = x + ((size_t)b * DIMC) * LSEQ + l;
  float v[6], s = 0.f, s2 = 0.f;
  #pragma unroll
  for (int j = 0; j < 6; ++j) {
    v[j] = xp[(size_t)(lane + j * 32) * LSEQ];
    s += v[j]; s2 += v[j] * v[j];
  }
  #pragma unroll
  for (int off = 16; off >= 1; off >>= 1) {
    s  += __shfl_xor(s,  off, 32);
    s2 += __shfl_xor(s2, off, 32);
  }
  float mu  = s * (1.f / DIMC);
  float var = s2 * (1.f / DIMC) - mu * mu;
  float rs  = rsqrtf(var + 1e-5f);
  #pragma unroll
  for (int j = 0; j < 6; ++j) {
    int c = lane + j * 32;
    XN[(size_t)row * DIMC + c] = (__bf16)((v[j] - mu) * rs * lnw[c] + lnb[c]);
  }
}

// ---------------------------------------------------------------------------
// Causal depthwise conv1d (taps=4) + SiLU, sequence-space output.
// rev=1 reads the spatial sequence reversed (direction-1 Mamba).
// ---------------------------------------------------------------------------
__global__ __launch_bounds__(256) void k_conv_silu(
    const float* __restrict__ XZ /*[M,384] (xm = cols 0..191)*/,
    const float* __restrict__ cw /*(192,4)*/, const float* __restrict__ cb,
    __bf16* __restrict__ XC /*[M,192] seq-space*/, int rev)
{
  int t = blockIdx.x * 256 + threadIdx.x;
  if (t >= BSZ * DIMC * (LSEQ / 8)) return;
  int c = t % DIMC; int tmp = t / DIMC;
  int chunk = tmp & (LSEQ / 8 - 1); int b = tmp / (LSEQ / 8);
  float w0 = cw[c * 4], w1 = cw[c * 4 + 1], w2 = cw[c * 4 + 2], w3 = cw[c * 4 + 3];
  float bias = cb[c];
  size_t rb = (size_t)b * LSEQ;
  int s0 = chunk * 8;
  float win[3];
  #pragma unroll
  for (int i = 0; i < 3; ++i) {
    int s = s0 - 3 + i;
    win[i] = (s >= 0) ? XZ[(rb + (rev ? LSEQ - 1 - s : s)) * DWDIM + c] : 0.f;
  }
  #pragma unroll
  for (int i = 0; i < 8; ++i) {
    int s = s0 + i;
    float cur = XZ[(rb + (rev ? LSEQ - 1 - s : s)) * DWDIM + c];
    float a = bias + w0 * win[0] + w1 * win[1] + w2 * win[2] + w3 * cur;
    float r = a / (1.f + expf(-a));                     // SiLU
    XC[(rb + s) * DIMC + c] = (__bf16)r;
    win[0] = win[1]; win[1] = win[2]; win[2] = cur;
  }
}

// ---------------------------------------------------------------------------
// Selective scan. 16 lanes = one (b,channel); lane = state index n (wave32 ->
// 2 channels/wave). Reduction over N via shfl_xor 8/4/2/1. Output is gated
// (y + D*xm) * silu(z) written in POSITION space (handles direction flip).
// ---------------------------------------------------------------------------
__global__ __launch_bounds__(256) void k_scan(
    const float* __restrict__ XD /*[M,224] seq: dt|B|C*/,
    const float* __restrict__ XZ /*[M,384] pos: z = cols 192..383*/,
    const __bf16* __restrict__ XC /*[M,192] seq*/,
    const float* __restrict__ dt_b, const float* __restrict__ A_log,
    const float* __restrict__ Dp, __bf16* __restrict__ GY /*[M,192] pos*/, int rev)
{
  int slot = blockIdx.x * 16 + (threadIdx.x >> 4);   // 0..383  (b*192+d)
  int n = threadIdx.x & 15;
  int b = slot / DIMC, d = slot % DIMC;
  float A   = -expf(A_log[d * NSTATE + n]);
  float dtb = dt_b[d];
  float Dd  = Dp[d];
  float h   = 0.f;
  size_t rb = (size_t)b * LSEQ;
  for (int s = 0; s < LSEQ; ++s) {
    size_t row  = rb + s;
    size_t prow = rb + (rev ? (LSEQ - 1 - s) : s);
    float dtp = XD[row * NCAT + d] + dtb;
    float dt  = dtp > 20.f ? dtp : log1pf(expf(dtp));   // softplus
    float xcv = (float)XC[row * DIMC + d];
    float Bv  = XD[row * NCAT + 192 + n];
    float Cv  = XD[row * NCAT + 208 + n];
    h = expf(dt * A) * h + dt * Bv * xcv;
    float p = h * Cv;
    p += __shfl_xor(p, 8, 32);
    p += __shfl_xor(p, 4, 32);
    p += __shfl_xor(p, 2, 32);
    p += __shfl_xor(p, 1, 32);
    if (n == 0) {
      float y = p + Dd * xcv;
      float z = XZ[prow * DWDIM + 192 + d];
      float g = y * (z / (1.f + expf(-z)));
      GY[prow * DIMC + d] = (__bf16)g;
    }
  }
}

// OUTM * scale -> XM (f32, kept for residual) and XMb (bf16 GEMM input)
__global__ void k_scale_cvt(const float* __restrict__ OUTM, const float* __restrict__ scale,
                            float* __restrict__ XM, __bf16* __restrict__ XMb, int n) {
  int i = blockIdx.x * 256 + threadIdx.x;
  if (i < n) { float v = OUTM[i] * scale[0]; XM[i] = v; XMb[i] = (__bf16)v; }
}

// Depthwise 3x3 (pad 1) on T[M,768] + bias, then gate: sg = t1 * sigmoid(t2)
__global__ __launch_bounds__(256) void k_dwconv_gate(
    const float* __restrict__ T, const float* __restrict__ dww /*(768,1,3,3)*/,
    const float* __restrict__ dwb, __bf16* __restrict__ SG /*[M,384]*/)
{
  int t = blockIdx.x * 256 + threadIdx.x;
  if (t >= MROWS * DWDIM) return;
  int c = t % DWDIM; int m = t / DWDIM;
  int b = m >> 12; int hw = m & 4095; int h = hw >> 6; int w = hw & 63;
  float v1 = dwb[c], v2 = dwb[c + DWDIM];
  #pragma unroll
  for (int ky = 0; ky < 3; ++ky) {
    int hh = h + ky - 1;
    if (hh < 0 || hh > 63) continue;
    #pragma unroll
    for (int kx = 0; kx < 3; ++kx) {
      int ww = w + kx - 1;
      if (ww < 0 || ww > 63) continue;
      size_t mm = ((size_t)b << 12) + hh * 64 + ww;
      const float* tp = T + mm * (2 * DWDIM);
      float wt1 = dww[c * 9 + ky * 3 + kx];
      float wt2 = dww[(c + DWDIM) * 9 + ky * 3 + kx];
      v1 += tp[c] * wt1;
      v2 += tp[c + DWDIM] * wt2;
    }
  }
  float sg = v1 * (1.f / (1.f + expf(-v2)));
  SG[(size_t)m * DWDIM + c] = (__bf16)sg;
}

// out = x + gamma * (XM + SGO)   (x layout B,C,H,W; XM/SGO are [M,192])
__global__ void k_final(const float* __restrict__ x, const float* __restrict__ XM,
                        const float* __restrict__ SGO, const float* __restrict__ gamma,
                        float* __restrict__ out) {
  int i = blockIdx.x * 256 + threadIdx.x;
  if (i >= BSZ * DIMC * LSEQ) return;
  int l = i & 4095; int tmp = i >> 12; int c = tmp % DIMC; int b = tmp / DIMC;
  size_t m = ((size_t)b << 12) + l;
  out[i] = x[i] + gamma[0] * (XM[m * DIMC + c] + SGO[m * DIMC + c]);
}

// ---------------------------------------------------------------------------
extern "C" void kernel_launch(void* const* d_in, const int* in_sizes, int n_in,
                              void* d_out, int out_size, void* d_ws, size_t ws_size,
                              hipStream_t stream) {
  (void)in_sizes; (void)n_in; (void)out_size; (void)ws_size;
  const float* x      = (const float*)d_in[0];
  const float* ln_w   = (const float*)d_in[1];
  const float* ln_b   = (const float*)d_in[2];
  const float* scale  = (const float*)d_in[3];
  const float* c1w    = (const float*)d_in[4];
  const float* c1b    = (const float*)d_in[5];
  const float* dww    = (const float*)d_in[6];
  const float* dwb    = (const float*)d_in[7];
  const float* c2w    = (const float*)d_in[8];
  const float* c2b    = (const float*)d_in[9];
  const float* gamma  = (const float*)d_in[10];
  const float* in_w[2]    = {(const float*)d_in[11], (const float*)d_in[20]};
  const float* conv_w[2]  = {(const float*)d_in[12], (const float*)d_in[21]};
  const float* conv_b[2]  = {(const float*)d_in[13], (const float*)d_in[22]};
  const float* xproj[2]   = {(const float*)d_in[14], (const float*)d_in[23]};
  const float* dt_w[2]    = {(const float*)d_in[15], (const float*)d_in[24]};
  const float* dt_b[2]    = {(const float*)d_in[16], (const float*)d_in[25]};
  const float* A_log[2]   = {(const float*)d_in[17], (const float*)d_in[26]};
  const float* Dp[2]      = {(const float*)d_in[18], (const float*)d_in[27]};
  const float* out_w[2]   = {(const float*)d_in[19], (const float*)d_in[28]};

  // ---- workspace layout (all 256B aligned) ----
  char* base = (char*)d_ws;
  size_t o = 0;
  auto take = [&](size_t bytes) { size_t r = o; o += (bytes + 255) & ~(size_t)255; return r; };
  __bf16* Win[2]; __bf16* Wx[2]; __bf16* Wout[2];
  Win[0]  = (__bf16*)(base + take(2 * DIMC * DIMC * 2 * sizeof(__bf16)) ); // 384x192 -> size computed below; keep explicit:
  o = 0;                                                                   // redo cleanly
  Win[0]  = (__bf16*)(base + take((size_t)DWDIM * DIMC * sizeof(__bf16)));
  Win[1]  = (__bf16*)(base + take((size_t)DWDIM * DIMC * sizeof(__bf16)));
  Wx[0]   = (__bf16*)(base + take((size_t)NCAT  * DIMC * sizeof(__bf16)));
  Wx[1]   = (__bf16*)(base + take((size_t)NCAT  * DIMC * sizeof(__bf16)));
  Wout[0] = (__bf16*)(base + take((size_t)DIMC  * DIMC * sizeof(__bf16)));
  Wout[1] = (__bf16*)(base + take((size_t)DIMC  * DIMC * sizeof(__bf16)));
  __bf16* Wc1 = (__bf16*)(base + take((size_t)(2 * DWDIM) * DIMC * sizeof(__bf16)));
  __bf16* Wc2 = (__bf16*)(base + take((size_t)DIMC * DWDIM * sizeof(__bf16)));
  __bf16* XN  = (__bf16*)(base + take((size_t)MROWS * DIMC * sizeof(__bf16)));
  size_t xz_off = take((size_t)MROWS * DWDIM * sizeof(float));  // XZ0
  float* XZ[2];
  XZ[0] = (float*)(base + xz_off);
  XZ[1] = (float*)(base + take((size_t)MROWS * DWDIM * sizeof(float)));
  float* T = (float*)(base + xz_off);                 // alias: T[M,768] reuses XZ0+XZ1
  __bf16* XC[2];
  XC[0] = (__bf16*)(base + take((size_t)MROWS * DIMC * sizeof(__bf16)));
  XC[1] = (__bf16*)(base + take((size_t)MROWS * DIMC * sizeof(__bf16)));
  size_t xd0_off = take((size_t)MROWS * NCAT * sizeof(float));
  size_t xd1_off = take((size_t)MROWS * NCAT * sizeof(float));
  float* XD[2] = {(float*)(base + xd0_off), (float*)(base + xd1_off)};
  __bf16* SG   = (__bf16*)(base + xd0_off);           // alias (XD dead after scan)
  float*  SGO  = (float*)(base + xd1_off);            // alias
  __bf16* GY[2];
  GY[0] = (__bf16*)(base + take((size_t)MROWS * DIMC * sizeof(__bf16)));
  GY[1] = (__bf16*)(base + take((size_t)MROWS * DIMC * sizeof(__bf16)));
  float* OUTM = (float*)(base + take((size_t)MROWS * DIMC * sizeof(float)));
  float* XM   = (float*)(base + take((size_t)MROWS * DIMC * sizeof(float)));
  __bf16* XMb = (__bf16*)(base + take((size_t)MROWS * DIMC * sizeof(__bf16)));

  auto cvt = [&](const float* s, __bf16* d, int n) {
    k_cvt<<<(n + 255) / 256, 256, 0, stream>>>(s, d, n);
  };
  auto gemm = [&](const __bf16* A, const __bf16* W, float* C, int M, int N, int K,
                  const float* bias, int acc) {
    dim3 grid(M / 128, N / 32);
    k_gemm_wmma<<<grid, 128, 0, stream>>>(A, W, C, M, N, K, bias, acc);
  };

  // 1) weight prep (f32 -> bf16, Δt-fold into concatenated projection)
  for (int d = 0; d < 2; ++d) {
    cvt(in_w[d],  Win[d],  DWDIM * DIMC);
    cvt(out_w[d], Wout[d], DIMC * DIMC);
    k_xcat<<<(NCAT * DIMC + 255) / 256, 256, 0, stream>>>(xproj[d], dt_w[d], Wx[d]);
  }
  cvt(c1w, Wc1, 2 * DWDIM * DIMC);
  cvt(c2w, Wc2, DIMC * DWDIM);

  // 2) LayerNorm -> bf16 rows
  k_layernorm<<<MROWS / 4, 128, 0, stream>>>(x, ln_w, ln_b, XN);

  // 3) in_proj (both directions share rows; reversal handled downstream)
  gemm(XN, Win[0], XZ[0], MROWS, DWDIM, DIMC, nullptr, 0);
  gemm(XN, Win[1], XZ[1], MROWS, DWDIM, DIMC, nullptr, 0);

  // 4) causal depthwise conv + SiLU (sequence space)
  const int convT = BSZ * DIMC * (LSEQ / 8);
  k_conv_silu<<<(convT + 255) / 256, 256, 0, stream>>>(XZ[0], conv_w[0], conv_b[0], XC[0], 0);
  k_conv_silu<<<(convT + 255) / 256, 256, 0, stream>>>(XZ[1], conv_w[1], conv_b[1], XC[1], 1);

  // 5) fused x_proj + Δt projection
  gemm(XC[0], Wx[0], XD[0], MROWS, NCAT, DIMC, nullptr, 0);
  gemm(XC[1], Wx[1], XD[1], MROWS, NCAT, DIMC, nullptr, 0);

  // 6) selective scan + gating (writes position space)
  k_scan<<<(BSZ * DIMC) / 16, 256, 0, stream>>>(XD[0], XZ[0], XC[0], dt_b[0], A_log[0], Dp[0], GY[0], 0);
  k_scan<<<(BSZ * DIMC) / 16, 256, 0, stream>>>(XD[1], XZ[1], XC[1], dt_b[1], A_log[1], Dp[1], GY[1], 1);

  // 7) out_proj, both directions summed
  gemm(GY[0], Wout[0], OUTM, MROWS, DIMC, DIMC, nullptr, 0);
  gemm(GY[1], Wout[1], OUTM, MROWS, DIMC, DIMC, nullptr, 1);

  // 8) scale_mod, keep f32 for residual + bf16 for SGFN GEMM
  k_scale_cvt<<<(MROWS * DIMC + 255) / 256, 256, 0, stream>>>(OUTM, scale, XM, XMb, MROWS * DIMC);

  // 9) SGFN: c1 (bias), dw3x3+gate, c2 (bias)
  gemm(XMb, Wc1, T, MROWS, 2 * DWDIM, DIMC, c1b, 0);
  k_dwconv_gate<<<(MROWS * DWDIM + 255) / 256, 256, 0, stream>>>(T, dww, dwb, SG);
  gemm(SG, Wc2, SGO, MROWS, DIMC, DWDIM, c2b, 0);

  // 10) residual
  k_final<<<(BSZ * DIMC * LSEQ + 255) / 256, 256, 0, stream>>>(x, XM, SGO, gamma, (float*)d_out);
}
